// AttentionModelCharLevel_60026462929172
// MI455X (gfx1250) — compile-verified, matrix operand order checked by reading, not run on hardware
//
#include <hip/hip_runtime.h>
#include <hip/hip_bf16.h>

// ---------------- problem constants ----------------
constexpr int NB = 4096;   // batch
constexpr int SL = 16;     // seq len
constexpr int HD = 512;    // hidden
constexpr int ED = 50;     // embed dim
constexpr int VD = 2048;   // visual dim
constexpr int NV = 128;    // vocab
constexpr int NG = 2048;   // 4*HD (gates)
constexpr int KC = 576;    // concat K (512 + 50, padded to mult of 32)

// ---------------- WMMA types ----------------
typedef __attribute__((ext_vector_type(16))) __bf16 v16bf;
typedef __attribute__((ext_vector_type(8)))  float  v8f;

__device__ __forceinline__ unsigned short f2bf(float f) {
  union { float f; unsigned u; } v; v.f = f;
  unsigned r = v.u + 0x7FFFu + ((v.u >> 16) & 1u);
  return (unsigned short)(r >> 16);
}
__device__ __forceinline__ float sigm(float x) { return 1.f / (1.f + __expf(-x)); }

// =====================================================================
// bf16 WMMA GEMM:  C[M,N](f32) = A[M,K](bf16,row-major) * Bn[N,K](bf16)^T
// (Bn is the weight in native [out,in] layout; C = A @ Bn^T)
// 128x128 tile per block, 256 threads = 8 waves (2 along M x 4 along N),
// each wave 64x32 of C = 4x2 tiles of v_wmma_f32_16x16x32_bf16.
// Double-buffered LDS, one barrier per K-step.
// Requires M%128==0, N%128==0, K%32==0.
// =====================================================================
#define TBM 128
#define TBN 128
#define TBK 32

template <int BIAS_RELU>
__global__ __launch_bounds__(256)
void gemm_bf16_wmma(const unsigned short* __restrict__ A,
                    const unsigned short* __restrict__ Bn,
                    float* __restrict__ C,
                    const float* __restrict__ bias,
                    int M, int N, int K)
{
  __shared__ __align__(32) unsigned short As[2][TBM * TBK];   // [m][k]
  __shared__ __align__(32) unsigned short Bs[2][TBN * TBK];   // [n][k]

  const int tid  = threadIdx.x;
  const int wave = tid >> 5;
  const int lane = tid & 31;
  const int m0 = blockIdx.y * TBM;
  const int n0 = blockIdx.x * TBN;
  const int wm = (wave >> 2) * 64;      // 2 waves along M
  const int wn = (wave & 3) * 32;       // 4 waves along N
  const int half = lane >> 4;
  const int lr   = lane & 15;

  // staging: tile is 128x32 = 4096 elems = 512 uint4; 256 threads -> 2 each
  const int r0  = (tid + 0)   >> 2;     // row for vector 0
  const int r1  = (tid + 256) >> 2;     // row for vector 1
  const int cg0 = ((tid + 0)   & 3) * 8;
  const int cg1 = ((tid + 256) & 3) * 8;

  uint4 ra0, ra1, rb0, rb1;
  auto stage_regs = [&](int k0) {
    ra0 = *(const uint4*)(A  + (size_t)(m0 + r0) * K + k0 + cg0);
    ra1 = *(const uint4*)(A  + (size_t)(m0 + r1) * K + k0 + cg1);
    rb0 = *(const uint4*)(Bn + (size_t)(n0 + r0) * K + k0 + cg0);
    rb1 = *(const uint4*)(Bn + (size_t)(n0 + r1) * K + k0 + cg1);
  };
  auto store_lds = [&](int buf) {
    *(uint4*)(&As[buf][r0 * TBK + cg0]) = ra0;
    *(uint4*)(&As[buf][r1 * TBK + cg1]) = ra1;
    *(uint4*)(&Bs[buf][r0 * TBK + cg0]) = rb0;
    *(uint4*)(&Bs[buf][r1 * TBK + cg1]) = rb1;
  };

  v8f acc[4][2] = {};

  stage_regs(0);
  store_lds(0);
  __syncthreads();

  int buf = 0;
  for (int k0 = 0; k0 < K; k0 += TBK) {
    const bool more = (k0 + TBK) < K;
    if (more) stage_regs(k0 + TBK);     // global loads overlap the WMMAs

    v16bf bfrag[2];
    #pragma unroll
    for (int j = 0; j < 2; ++j)
      bfrag[j] = *(const v16bf*)(const void*)
          (&Bs[buf][(wn + j * 16 + lr) * TBK + half * 16]);
    #pragma unroll
    for (int i = 0; i < 4; ++i) {
      v16bf a = *(const v16bf*)(const void*)
          (&As[buf][(wm + i * 16 + lr) * TBK + half * 16]);
      #pragma unroll
      for (int j = 0; j < 2; ++j)
        acc[i][j] = __builtin_amdgcn_wmma_f32_16x16x32_bf16(
            false, a, false, bfrag[j], (short)0, acc[i][j], false, false);
    }

    if (more) {
      store_lds(buf ^ 1);
      __syncthreads();
      buf ^= 1;
    }
  }

  // epilogue: lane L, vgpr r -> row = r + 8*half, col = lr (per 16x16 tile)
  #pragma unroll
  for (int i = 0; i < 4; ++i)
    #pragma unroll
    for (int j = 0; j < 2; ++j)
      #pragma unroll
      for (int r = 0; r < 8; ++r) {
        int row = m0 + wm + i * 16 + r + 8 * half;
        int col = n0 + wn + j * 16 + lr;
        float x = acc[i][j][r];
        if (BIAS_RELU) {
          x += bias[col];
          x = fmaxf(x, 0.f);
        }
        C[(size_t)row * N + col] = x;
      }
}

// =====================================================================
// small elementwise / prep kernels
// =====================================================================
__global__ void lengths_kernel(const int* __restrict__ text, int* __restrict__ len) {
  int b = blockIdx.x * blockDim.x + threadIdx.x;
  if (b >= NB) return;
  int n = 0;
  for (int l = 0; l < SL; ++l) n += (text[b * SL + l] != 0);
  len[b] = n;
}

// Wcat[g*KC + k] = k<HD ? W_hh[g,k] : k<HD+ED ? W_ih[g,k-HD] : 0   (native [N,K])
__global__ void build_gatesW_kernel(const float* __restrict__ W_hh,
                                    const float* __restrict__ W_ih,
                                    unsigned short* __restrict__ Wm) {
  int idx = blockIdx.x * blockDim.x + threadIdx.x;
  if (idx >= NG * KC) return;
  int g = idx / KC, k = idx % KC;
  float v = 0.f;
  if (k < HD)           v = W_hh[(size_t)g * HD + k];
  else if (k < HD + ED) v = W_ih[(size_t)g * ED + (k - HD)];
  Wm[idx] = f2bf(v);
}

__global__ void cvt_bf16_kernel(const float* __restrict__ s,
                                unsigned short* __restrict__ d, int n) {
  int i = blockIdx.x * blockDim.x + threadIdx.x;
  if (i < n) d[i] = f2bf(s[i]);
}
__global__ void fill_f32_kernel(float* p, float v, int n) {
  int i = blockIdx.x * blockDim.x + threadIdx.x;
  if (i < n) p[i] = v;
}
__global__ void fill_u16_kernel(unsigned short* p, unsigned short v, int n) {
  int i = blockIdx.x * blockDim.x + threadIdx.x;
  if (i < n) p[i] = v;
}

// xcat[b, 0:512] = bf16(h[b,:]); xcat[b, 512:562] = bf16(emb[char]); pad stays 0
__global__ void prep_xcat_kernel(unsigned short* __restrict__ xcat,
                                 const float* __restrict__ h,
                                 const float* __restrict__ emb,
                                 const int* __restrict__ text,
                                 int t, int dec) {
  int idx = blockIdx.x * blockDim.x + threadIdx.x;
  if (idx >= NB * KC) return;
  int b = idx / KC, j = idx % KC;
  if (j < HD) {
    xcat[idx] = f2bf(h[(size_t)b * HD + j]);
  } else if (j < HD + ED) {
    int ch = dec ? ((t == 0) ? 0 : text[b * SL + t - 1]) : text[b * SL + t];
    xcat[idx] = f2bf(emb[(size_t)ch * ED + (j - HD)]);
  }
}

// gate order i,f,g,o (columns 0..511, 512..1023, 1024..1535, 1536..2047)
__global__ void lstm_cell_kernel(const float* __restrict__ gates,
                                 const float* __restrict__ b_ih,
                                 const float* __restrict__ b_hh,
                                 float* __restrict__ h, float* __restrict__ c,
                                 const int* __restrict__ len, int t,
                                 unsigned short* __restrict__ tenc_bf,  // nullable
                                 unsigned short* __restrict__ hbf) {    // nullable
  int idx = blockIdx.x * blockDim.x + threadIdx.x;
  if (idx >= NB * HD) return;
  int b = idx / HD, j = idx % HD;
  const float* g = gates + (size_t)b * NG;
  float ig = g[j]          + b_ih[j]          + b_hh[j];
  float fg = g[HD + j]     + b_ih[HD + j]     + b_hh[HD + j];
  float gg = g[2 * HD + j] + b_ih[2 * HD + j] + b_hh[2 * HD + j];
  float og = g[3 * HD + j] + b_ih[3 * HD + j] + b_hh[3 * HD + j];
  float cs = sigm(fg) * c[idx] + sigm(ig) * tanhf(gg);
  float hs = sigm(og) * tanhf(cs);
  c[idx] = cs;
  h[idx] = hs;
  if (tenc_bf && (len[b] - 1 == t)) tenc_bf[idx] = f2bf(hs);
  if (hbf) hbf[idx] = f2bf(hs);
}

// row L2-normalize src[NB,HD]; optional bf16, transposed bf16, two f32 dests
__global__ void rownorm_kernel(const float* __restrict__ src,
                               unsigned short* __restrict__ dst_bf,
                               unsigned short* __restrict__ dstT_bf,
                               float* __restrict__ dst_a,
                               float* __restrict__ dst_b) {
  int b = blockIdx.x, tid = threadIdx.x;      // 128 threads
  __shared__ float red[128];
  float ss = 0.f;
  for (int j = tid; j < HD; j += 128) {
    float x = src[(size_t)b * HD + j];
    ss += x * x;
  }
  red[tid] = ss; __syncthreads();
  for (int s = 64; s > 0; s >>= 1) {
    if (tid < s) red[tid] += red[tid + s];
    __syncthreads();
  }
  float rn = rsqrtf(red[0]);
  for (int j = tid; j < HD; j += 128) {
    float x = src[(size_t)b * HD + j] * rn;
    if (dst_bf)  dst_bf[(size_t)b * HD + j] = f2bf(x);
    if (dstT_bf) dstT_bf[(size_t)j * NB + b] = f2bf(x);
    if (dst_a)   dst_a[(size_t)b * HD + j] = x;
    if (dst_b)   dst_b[(size_t)b * HD + j] = x;
  }
}

// row softmax of simsT[NB,NB] -> P bf16 (row j of simsT == column j of sims)
__global__ void softmax_row_kernel(const float* __restrict__ S,
                                   unsigned short* __restrict__ P) {
  int row = blockIdx.x, tid = threadIdx.x;    // 256 threads
  const float* x = S + (size_t)row * NB;
  __shared__ float red[256];
  float m = -1e30f;
  for (int j = tid; j < NB; j += 256) m = fmaxf(m, x[j]);
  red[tid] = m; __syncthreads();
  for (int s = 128; s > 0; s >>= 1) {
    if (tid < s) red[tid] = fmaxf(red[tid], red[tid + s]);
    __syncthreads();
  }
  m = red[0]; __syncthreads();
  float sum = 0.f;
  for (int j = tid; j < NB; j += 256) sum += __expf(x[j] - m);
  red[tid] = sum; __syncthreads();
  for (int s = 128; s > 0; s >>= 1) {
    if (tid < s) red[tid] += red[tid + s];
    __syncthreads();
  }
  float inv = 1.f / red[0];
  for (int j = tid; j < NB; j += 256)
    P[(size_t)row * NB + j] = f2bf(__expf(x[j] - m) * inv);
}

// per-row log-softmax NLL over NV=128 logits, accumulate into loss[b]
__global__ void dec_loss_kernel(const float* __restrict__ logits,
                                const int* __restrict__ text,
                                float* __restrict__ loss, int t) {
  int b = blockIdx.x, tid = threadIdx.x;      // 128 threads
  __shared__ float red[128];
  float x = logits[(size_t)b * NV + tid];
  red[tid] = x; __syncthreads();
  for (int s = 64; s > 0; s >>= 1) {
    if (tid < s) red[tid] = fmaxf(red[tid], red[tid + s]);
    __syncthreads();
  }
  float m = red[0]; __syncthreads();
  red[tid] = __expf(x - m); __syncthreads();
  for (int s = 64; s > 0; s >>= 1) {
    if (tid < s) red[tid] += red[tid + s];
    __syncthreads();
  }
  if (tid == 0) {
    int tgt = text[b * SL + t];
    float lp = logits[(size_t)b * NV + tgt] - m - __logf(red[0]);
    loss[b] -= lp;
  }
}

__global__ void finalize_kernel(const float* __restrict__ loss,
                                float* __restrict__ out, float inv) {
  int b = blockIdx.x * blockDim.x + threadIdx.x;
  if (b < NB) out[b] = loss[b] * inv;
}

// =====================================================================
// host side
// =====================================================================
static inline dim3 eltgrid(long n) { return dim3((unsigned)((n + 255) / 256)); }

extern "C" void kernel_launch(void* const* d_in, const int* in_sizes, int n_in,
                              void* d_out, int out_size, void* d_ws, size_t ws_size,
                              hipStream_t stream) {
  const float* visual = (const float*)d_in[0];
  const int*   text   = (const int*)d_in[1];
  const float* emb    = (const float*)d_in[2];
  const float* W_ih   = (const float*)d_in[3];
  const float* W_hh   = (const float*)d_in[4];
  const float* b_ih   = (const float*)d_in[5];
  const float* b_hh   = (const float*)d_in[6];
  const float* W_enc  = (const float*)d_in[7];
  const float* b_enc  = (const float*)d_in[8];
  const float* W_out  = (const float*)d_in[9];
  const float* W_vis  = (const float*)d_in[10];
  float* out = (float*)d_out;

  char* ws = (char*)d_ws;
  size_t off = 0;
  auto alloc = [&](size_t bytes) -> void* {
    void* p = ws + off;
    off += (bytes + 255) & ~(size_t)255;
    return p;
  };

  unsigned short* gatesW = (unsigned short*)alloc((size_t)NG * KC * 2); // [NG,KC]
  unsigned short* visW   = (unsigned short*)alloc((size_t)HD * VD * 2); // [HD,VD]
  unsigned short* encW   = (unsigned short*)alloc((size_t)HD * HD * 2); // [HD,HD]
  unsigned short* outW   = (unsigned short*)alloc((size_t)NV * HD * 2); // [NV,HD]
  unsigned short* visbf  = (unsigned short*)alloc((size_t)NB * VD * 2);
  unsigned short* xcat   = (unsigned short*)alloc((size_t)NB * KC * 2);
  float*          gates  = (float*)alloc((size_t)NB * NG * 4);
  float*          hbuf   = (float*)alloc((size_t)NB * HD * 4);
  float*          cbuf   = (float*)alloc((size_t)NB * HD * 4);
  float*          vbuf   = (float*)alloc((size_t)NB * HD * 4);
  float*          tlin   = (float*)alloc((size_t)NB * HD * 4);
  unsigned short* tencbf = (unsigned short*)alloc((size_t)NB * HD * 2);
  unsigned short* vnbf   = (unsigned short*)alloc((size_t)NB * HD * 2); // [NB,HD]
  unsigned short* vnT    = (unsigned short*)alloc((size_t)HD * NB * 2); // [HD,NB]
  unsigned short* tnbf   = (unsigned short*)alloc((size_t)NB * HD * 2);
  float*          simsT  = (float*)alloc((size_t)NB * NB * 4);
  unsigned short* Pmat   = (unsigned short*)alloc((size_t)NB * NB * 2);
  float*          hdec   = (float*)alloc((size_t)NB * HD * 4);
  unsigned short* hbf    = (unsigned short*)alloc((size_t)NB * HD * 2);
  float*          logits = (float*)alloc((size_t)NB * NV * 4);
  int*            len    = (int*)alloc((size_t)NB * 4);
  float*          loss   = (float*)alloc((size_t)NB * 4);

  auto gemm = [&](const unsigned short* A, const unsigned short* Bn, float* C,
                  const float* bias, int relu, int M, int N, int K) {
    dim3 g(N / TBN, M / TBM);
    if (relu)
      gemm_bf16_wmma<1><<<g, dim3(256), 0, stream>>>(A, Bn, C, bias, M, N, K);
    else
      gemm_bf16_wmma<0><<<g, dim3(256), 0, stream>>>(A, Bn, C, bias, M, N, K);
  };

  // ---- weight prep (every call; deterministic) ----
  lengths_kernel<<<eltgrid(NB), 256, 0, stream>>>(text, len);
  build_gatesW_kernel<<<eltgrid((long)NG * KC), 256, 0, stream>>>(W_hh, W_ih, gatesW);
  cvt_bf16_kernel<<<eltgrid((long)HD * VD), 256, 0, stream>>>(W_vis, visW, HD * VD);
  cvt_bf16_kernel<<<eltgrid((long)HD * HD), 256, 0, stream>>>(W_enc, encW, HD * HD);
  cvt_bf16_kernel<<<eltgrid((long)NV * HD), 256, 0, stream>>>(W_out, outW, NV * HD);
  cvt_bf16_kernel<<<eltgrid((long)NB * VD), 256, 0, stream>>>(visual, visbf, NB * VD);

  // ---- state init ----
  fill_u16_kernel<<<eltgrid((long)NB * KC), 256, 0, stream>>>(xcat, (unsigned short)0, NB * KC);
  fill_f32_kernel<<<eltgrid((long)NB * HD), 256, 0, stream>>>(hbuf, 0.1f, NB * HD);
  fill_f32_kernel<<<eltgrid((long)NB * HD), 256, 0, stream>>>(cbuf, 0.1f, NB * HD);
  fill_f32_kernel<<<eltgrid(NB), 256, 0, stream>>>(loss, 0.f, NB);

  // ---- visual projection: v = vis @ W_vis.T ----
  gemm(visbf, visW, vbuf, nullptr, 0, NB, HD, VD);

  // ---- encoder LSTM ----
  for (int t = 0; t < SL; ++t) {
    prep_xcat_kernel<<<eltgrid((long)NB * KC), 256, 0, stream>>>(xcat, hbuf, emb, text, t, 0);
    gemm(xcat, gatesW, gates, nullptr, 0, NB, NG, KC);
    lstm_cell_kernel<<<eltgrid((long)NB * HD), 256, 0, stream>>>(
        gates, b_ih, b_hh, hbuf, cbuf, len, t, tencbf, nullptr);
  }

  // ---- enc linear + relu ----
  gemm(tencbf, encW, tlin, b_enc, 1, NB, HD, HD);

  // ---- normalize v, t ----
  rownorm_kernel<<<NB, 128, 0, stream>>>(vbuf, vnbf, vnT, nullptr, nullptr);
  rownorm_kernel<<<NB, 128, 0, stream>>>(tlin, tnbf, nullptr, nullptr, nullptr);

  // ---- simsT = t_n @ v_n^T (B-operand = vnbf in native [N,K]) ----
  gemm(tnbf, vnbf, simsT, nullptr, 0, NB, NB, HD);
  softmax_row_kernel<<<NB, 256, 0, stream>>>(simsT, Pmat);
  // ---- h = P @ v_n (B-operand = vnT [HD,NB]) ----
  gemm(Pmat, vnT, hdec, nullptr, 0, NB, HD, NB);

  // ---- normalize h -> decoder h0 = c0 ----
  rownorm_kernel<<<NB, 128, 0, stream>>>(hdec, nullptr, nullptr, hbuf, cbuf);

  // ---- decoder LSTM + per-step NLL ----
  for (int t = 0; t < SL; ++t) {
    prep_xcat_kernel<<<eltgrid((long)NB * KC), 256, 0, stream>>>(xcat, hbuf, emb, text, t, 1);
    gemm(xcat, gatesW, gates, nullptr, 0, NB, NG, KC);
    lstm_cell_kernel<<<eltgrid((long)NB * HD), 256, 0, stream>>>(
        gates, b_ih, b_hh, hbuf, cbuf, len, t, nullptr, hbf);
    gemm(hbf, outW, logits, nullptr, 0, NB, NV, HD);
    dec_loss_kernel<<<NB, 128, 0, stream>>>(logits, text, loss, t);
  }

  finalize_kernel<<<eltgrid(NB), 256, 0, stream>>>(loss, out, 1.0f / SL);
  (void)in_sizes; (void)n_in; (void)out_size; (void)ws_size;
}
